// SparseTopKAttention_16569983828646
// MI455X (gfx1250) — matrix operand order repaired and checked by reading, not compiled, not run on hardware
//
#include <hip/hip_runtime.h>

// ---------------------------------------------------------------------------
// SparseTopKAttention for MI455X (gfx1250, wave32, WMMA + TDM).
// bf16 WMMA (f32 accumulate) for dense GEMMs + QK^T scores; K-panel staged
// into LDS by the Tensor Data Mover (double-buffered, s_wait_tensorcnt);
// fused top-k/softmax/ctx with 128KB LDS score panel; f32 atomics head-mean.
// ---------------------------------------------------------------------------

typedef __attribute__((ext_vector_type(16))) __bf16 v16bf;
typedef __attribute__((ext_vector_type(8)))  float  v8f;
typedef __attribute__((ext_vector_type(4)))  unsigned int v4u;
typedef __attribute__((ext_vector_type(8)))  int v8i;
typedef __attribute__((ext_vector_type(4)))  int v4i;

union BVec { v16bf v; uint4 q[2]; };        // 16 bf16 = 32 bytes
union VAcc { v8f  v; float f[8]; };

__device__ __forceinline__ unsigned short f32_to_bf16(float f) {
  unsigned u = __float_as_uint(f);
  u += 0x7fffu + ((u >> 16) & 1u);          // round to nearest even
  return (unsigned short)(u >> 16);
}
__device__ __forceinline__ float bf16_to_f32(unsigned short h) {
  return __uint_as_float(((unsigned)h) << 16);
}

#define S_LEN   2048
#define E_DIM   1024
#define H_NUM   16
#define DH      64
#define TOPK    32
#define KSTAGE_BYTES (8 * 2 * 2048)         // 8 waves x 2 buffers x 16x64 bf16

// ---------------------------------------------------------------------------
// TDM: load one 16-row x 64-col bf16 tile (row stride 1024 elems) into LDS.
// D# packing per CDNA5 ISA 8.3/8.4: count=1, type=2(image), data_size=2B,
// tile_dim0=64, tile_dim1=16, tensor_dim0_stride=1024. Wave-uniform args.
// ---------------------------------------------------------------------------
__device__ __forceinline__ void tdm_load_k_tile(const unsigned short* gptr,
                                                unsigned lds_off) {
  const unsigned long long ga = (unsigned long long)(size_t)gptr;
  v4u g0;
  g0[0] = 1u;                                        // count=1, user mode
  g0[1] = lds_off;                                   // lds_addr (bytes)
  g0[2] = (unsigned)(ga & 0xffffffffu);              // global_addr[31:0]
  g0[3] = (unsigned)((ga >> 32) & 0x1ffffffu)        // global_addr[56:32]
        | (2u << 30);                                // type=2
  v8i g1;
  g1[0] = (1 << 16);            // workgroup_mask=0, data_size=1 (2 bytes)
  g1[1] = (64 << 16);           // tensor_dim0[15:0]=64 in bits [31:16]
  g1[2] = (16 << 16);           // tensor_dim0 hi=0 | tensor_dim1[15:0]=16
  g1[3] = (64 << 16);           // tensor_dim1 hi=0 | tile_dim0=64
  g1[4] = 16;                   // tile_dim1=16, tile_dim2=0
  g1[5] = 1024;                 // tensor_dim0_stride[31:0] (elements)
  g1[6] = 0;                    // stride hi | tensor_dim1_stride lo
  g1[7] = 0;
  v4i g2 = {0, 0, 0, 0};        // tensor_dim2.. unused (zeros)
  v4i g3 = {0, 0, 0, 0};
#if __clang_major__ >= 23
  v8i g4 = {0, 0, 0, 0, 0, 0, 0, 0};
  __builtin_amdgcn_tensor_load_to_lds(g0, g1, g2, g3, g4, 0);
#else
  __builtin_amdgcn_tensor_load_to_lds(g0, g1, g2, g3, 0);
#endif
}

// ---------------------------------------------------------------------------
// f32 -> bf16 conversion
// ---------------------------------------------------------------------------
__global__ void cvt_f32_bf16(const float* __restrict__ src,
                             unsigned short* __restrict__ dst, int n) {
  int i = blockIdx.x * blockDim.x + threadIdx.x;
  if (i < n) dst[i] = f32_to_bf16(src[i]);
}

// ---------------------------------------------------------------------------
// C[M,N] = A[M,K] @ W[N,K]^T + bias[N]   (NT GEMM, bf16 in, f32 acc)
// 8 waves/block; each wave owns a 16x32 C strip (1x2 register blocking:
// one A tile reused by two B tiles -> 2 wmma per 6 b128 loads).
// ---------------------------------------------------------------------------
__global__ void gemm_nt(const unsigned short* __restrict__ A,
                        const unsigned short* __restrict__ W,
                        const float* __restrict__ bias,
                        unsigned short* __restrict__ Cb,   // bf16 out (or null)
                        float* __restrict__ Cf,            // f32 out (or null)
                        int M, int N, int K) {
  const int lane  = threadIdx.x & 31;
  const int wave  = threadIdx.x >> 5;
  const int strip = blockIdx.x * 8 + wave;
  const int nstr  = N >> 5;                  // strips along N (32 cols each)
  const int tm    = strip / nstr;
  const int tn    = strip - tm * nstr;
  if (tm >= (M >> 4)) return;

  const int mrow  = (tm << 4) + (lane & 15);
  const int ncol0 = (tn << 5) + (lane & 15);
  const int ncol1 = ncol0 + 16;
  const int kbA   = (lane & 16) ? 8  : 0;
  const int kbB   = (lane & 16) ? 16 : 0;

  const unsigned short* arow  = A + (size_t)mrow  * K;
  const unsigned short* wrow0 = W + (size_t)ncol0 * K;
  const unsigned short* wrow1 = W + (size_t)ncol1 * K;

  VAcc acc0, acc1;
#pragma unroll
  for (int r = 0; r < 8; ++r) { acc0.f[r] = 0.0f; acc1.f[r] = 0.0f; }

  for (int kk = 0; kk < K; kk += 32) {
    BVec a, b0, b1;
    a.q[0]  = *(const uint4*)(arow  + kk + kbA);
    a.q[1]  = *(const uint4*)(arow  + kk + 16 + kbA);
    b0.q[0] = *(const uint4*)(wrow0 + kk + kbA);
    b0.q[1] = *(const uint4*)(wrow0 + kk + 16 + kbA);
    b1.q[0] = *(const uint4*)(wrow1 + kk + kbA);
    b1.q[1] = *(const uint4*)(wrow1 + kk + 16 + kbA);
    if (kk + 128 < K) {
      // loc=3 -> WGP-scope prefetch (fills all cache levels), ISA 10.5
      __builtin_prefetch(arow  + kk + 128, 0, 3);
      __builtin_prefetch(wrow0 + kk + 128, 0, 3);
      __builtin_prefetch(wrow1 + kk + 128, 0, 3);
    }
    acc0.v = __builtin_amdgcn_wmma_f32_16x16x32_bf16(
        false, a.v, false, b0.v, (short)0, acc0.v, false, false);
    acc1.v = __builtin_amdgcn_wmma_f32_16x16x32_bf16(
        false, a.v, false, b1.v, (short)0, acc1.v, false, false);
  }

  const int   mbase = (tm << 4) + ((lane & 16) ? 8 : 0);
  const float bn0   = bias[ncol0];
  const float bn1   = bias[ncol1];
#pragma unroll
  for (int r = 0; r < 8; ++r) {
    const size_t i0 = (size_t)(mbase + r) * N + ncol0;
    const size_t i1 = (size_t)(mbase + r) * N + ncol1;
    const float  v0 = acc0.f[r] + bn0;
    const float  v1 = acc1.f[r] + bn1;
    if (Cb) { Cb[i0] = f32_to_bf16(v0); Cb[i1] = f32_to_bf16(v1); }
    else    { Cf[i0] = v0;              Cf[i1] = v1; }
  }
}

// ---------------------------------------------------------------------------
// Fused attention: one block = (head h, 16 query rows).
//  - K tiles staged to LDS by TDM (double-buffered per wave, TENSORcnt)
//  - 8 waves compute scores (WMMA) into LDS panel sc[16][2048]
//  - per-row 32-pass argmax selection (jax tie-break: lower index wins)
//  - softmax over 32 survivors; scatter p/16 into head-mean via f32 atomics
//  - sparse ctx gather from V; ctx stored bf16 for the output projection
// LDS: [kstage 32KB][sc 128KB][topv|topi|pv 6KB] = 166KB (< 320KB/WGP)
// ---------------------------------------------------------------------------
__global__ void attn_topk_kernel(const unsigned short* __restrict__ Qb,
                                 const unsigned short* __restrict__ Kb,
                                 const unsigned short* __restrict__ Vb,
                                 unsigned short* __restrict__ ctxb,
                                 float* __restrict__ amean) {
  extern __shared__ char smem[];
  char*  kstage = smem;                                   // 32 KB, offset 0
  float* sc     = (float*)(smem + KSTAGE_BYTES);          // 16 * 2048 f32
  float* topv   = sc + 16 * S_LEN;
  int*   topi   = (int*)(topv + 16 * TOPK);
  float* pv     = (float*)(topi + 16 * TOPK);

  const int h  = blockIdx.x >> 7;                         // 16 heads
  const int q0 = (blockIdx.x & 127) << 4;                 // 128 q-blocks of 16

  const int lane = threadIdx.x & 31;
  const int wave = threadIdx.x >> 5;
  const float NEG_INF = __uint_as_float(0xff800000u);

  // ---- scores: S[q, c] = (1/8) * sum_d Q[q,d] * K[c,d] -------------------
  const int qrow = q0 + (lane & 15);
  const unsigned short* qp = Qb + (size_t)qrow * E_DIM + h * DH;
  const int kbA = (lane & 16) ? 8  : 0;
  const int kbB = (lane & 16) ? 16 : 0;

  BVec a0, a1;
  a0.q[0] = *(const uint4*)(qp + kbA);
  a0.q[1] = *(const uint4*)(qp + 16 + kbA);
  a1.q[0] = *(const uint4*)(qp + 32 + kbA);
  a1.q[1] = *(const uint4*)(qp + 48 + kbA);

  const int col = lane & 15;
  const unsigned sbase = (unsigned)(wave * 2) * 2048u;    // this wave's buffers
  const unsigned short* kpanel = Kb + (size_t)(wave * 256) * E_DIM + h * DH;

  // prime double buffer with tile 0
  tdm_load_k_tile(kpanel, sbase);

  for (int t = 0; t < 16; ++t) {
    if (t + 1 < 16) {
      tdm_load_k_tile(kpanel + (size_t)((t + 1) * 16) * E_DIM,
                      sbase + (unsigned)((t + 1) & 1) * 2048u);
      __builtin_amdgcn_s_wait_tensorcnt(1);   // tile t landed, t+1 in flight
    } else {
      __builtin_amdgcn_s_wait_tensorcnt(0);
    }

    const char* kl = kstage + sbase + (t & 1) * 2048 + col * 128;
    BVec b0, b1;
    b0.q[0] = *(const uint4*)(kl + kbA * 2);        // dh = kbA..kbA+7, 16+kbA..
    b0.q[1] = *(const uint4*)(kl + kbA * 2 + 32);
    b1.q[0] = *(const uint4*)(kl + 64 + kbA * 2);   // dh = 32+kbA.., 48+kbA..
    b1.q[1] = *(const uint4*)(kl + 64 + kbA * 2 + 32);

    VAcc acc;
#pragma unroll
    for (int r = 0; r < 8; ++r) acc.f[r] = 0.0f;
    acc.v = __builtin_amdgcn_wmma_f32_16x16x32_bf16(
        false, a0.v, false, b0.v, (short)0, acc.v, false, false);
    acc.v = __builtin_amdgcn_wmma_f32_16x16x32_bf16(
        false, a1.v, false, b1.v, (short)0, acc.v, false, false);

    const int c0    = wave * 256 + t * 16;
    const int mbase = (lane & 16) ? 8 : 0;
#pragma unroll
    for (int r = 0; r < 8; ++r)
      sc[(mbase + r) * S_LEN + c0 + col] = acc.f[r] * 0.125f;  // 1/sqrt(64)
  }
  __syncthreads();

  // ---- top-32 per row: 2 rows per wave, 32 selection passes --------------
  for (int rr = 0; rr < 2; ++rr) {
    const int row = wave * 2 + rr;
    for (int sel = 0; sel < TOPK; ++sel) {
      float best = NEG_INF;
      int   bidx = 0x7fffffff;
      for (int i = lane; i < S_LEN; i += 32) {
        const float v = sc[row * S_LEN + i];
        if (v > best) { best = v; bidx = i; }
      }
#pragma unroll
      for (int off = 16; off > 0; off >>= 1) {
        const float ov = __shfl_xor(best, off, 32);
        const int   oi = __shfl_xor(bidx, off, 32);
        if (ov > best || (ov == best && oi < bidx)) { best = ov; bidx = oi; }
      }
      if (lane == 0) {
        topv[row * TOPK + sel] = best;
        topi[row * TOPK + sel] = bidx;
        sc[row * S_LEN + bidx] = NEG_INF;      // remove from next pass
      }
      __syncthreads();                         // uniform across all 8 waves
    }
  }
  __syncthreads();

  // ---- softmax over 32 survivors + scatter mean-over-heads ---------------
  for (int rr = 0; rr < 2; ++rr) {
    const int row = wave * 2 + rr;
    const int q   = q0 + row;
    const float v = topv[row * TOPK + lane];   // all 32 lanes used
    float mx = v;
#pragma unroll
    for (int off = 16; off > 0; off >>= 1) mx = fmaxf(mx, __shfl_xor(mx, off, 32));
    const float e = __expf(v - mx);
    float s = e;
#pragma unroll
    for (int off = 16; off > 0; off >>= 1) s += __shfl_xor(s, off, 32);
    const float p = e / s;
    pv[row * TOPK + lane] = p;
    atomicAdd(&amean[(size_t)q * S_LEN + topi[row * TOPK + lane]],
              p * (1.0f / H_NUM));
  }
  __syncthreads();

  // ---- sparse ctx: ctx[q, h*64+d] = sum_j p_j * V[idx_j, h*64+d] ---------
  for (int rr = 0; rr < 2; ++rr) {
    const int row = wave * 2 + rr;
    const int q   = q0 + row;
    const int d0  = lane * 2;                  // 2 dh columns per lane
    float acc0 = 0.0f, acc1 = 0.0f;
    for (int j = 0; j < TOPK; ++j) {
      const float p   = pv[row * TOPK + j];
      const int   idx = topi[row * TOPK + j];
      const unsigned pk =
          *(const unsigned*)(Vb + (size_t)idx * E_DIM + h * DH + d0);
      acc0 += p * bf16_to_f32((unsigned short)(pk & 0xffffu));
      acc1 += p * bf16_to_f32((unsigned short)(pk >> 16));
    }
    const unsigned outpk = (unsigned)f32_to_bf16(acc0) |
                           ((unsigned)f32_to_bf16(acc1) << 16);
    *(unsigned*)(ctxb + (size_t)q * E_DIM + h * DH + d0) = outpk;
  }
}

// ---------------------------------------------------------------------------
// Host launcher
// ---------------------------------------------------------------------------
extern "C" void kernel_launch(void* const* d_in, const int* in_sizes, int n_in,
                              void* d_out, int out_size, void* d_ws, size_t ws_size,
                              hipStream_t stream) {
  const float* query = (const float*)d_in[0];
  const float* key   = (const float*)d_in[1];
  const float* value = (const float*)d_in[2];
  const float* wq    = (const float*)d_in[3];
  const float* bq    = (const float*)d_in[4];
  const float* wk    = (const float*)d_in[5];
  const float* bk    = (const float*)d_in[6];
  const float* wv    = (const float*)d_in[7];
  const float* bv    = (const float*)d_in[8];
  const float* wo    = (const float*)d_in[9];
  const float* bo    = (const float*)d_in[10];

  const int SE = S_LEN * E_DIM;   // 2 M elements
  const int EE = E_DIM * E_DIM;   // 1 M elements

  // workspace layout (bf16 halves), total 18M elems = 36 MB
  unsigned short* ws  = (unsigned short*)d_ws;
  unsigned short* qx  = ws;            // query  bf16
  unsigned short* kx  = qx  + SE;      // key    bf16
  unsigned short* vx  = kx  + SE;      // value  bf16
  unsigned short* wqb = vx  + SE;
  unsigned short* wkb = wqb + EE;
  unsigned short* wvb = wkb + EE;
  unsigned short* wob = wvb + EE;
  unsigned short* Qb  = wob + EE;      // Q projection bf16
  unsigned short* Kb  = Qb  + SE;
  unsigned short* Vb  = Kb  + SE;
  unsigned short* ctx = Vb  + SE;      // context bf16

  float* out   = (float*)d_out;                 // [2048, 1024]
  float* amean = out + (size_t)SE;              // [2048, 2048]

  // zero the atomic-accumulated head-mean output
  hipMemsetAsync(amean, 0, (size_t)S_LEN * S_LEN * sizeof(float), stream);

  // f32 -> bf16 conversions
  cvt_f32_bf16<<<(SE + 255) / 256, 256, 0, stream>>>(query, qx, SE);
  cvt_f32_bf16<<<(SE + 255) / 256, 256, 0, stream>>>(key,   kx, SE);
  cvt_f32_bf16<<<(SE + 255) / 256, 256, 0, stream>>>(value, vx, SE);
  cvt_f32_bf16<<<(EE + 255) / 256, 256, 0, stream>>>(wq, wqb, EE);
  cvt_f32_bf16<<<(EE + 255) / 256, 256, 0, stream>>>(wk, wkb, EE);
  cvt_f32_bf16<<<(EE + 255) / 256, 256, 0, stream>>>(wv, wvb, EE);
  cvt_f32_bf16<<<(EE + 255) / 256, 256, 0, stream>>>(wo, wob, EE);

  // Q/K/V projections: (M/16)*(N/32) strips / 8 waves = 512 blocks
  const int gblocks = (S_LEN / 16) * (E_DIM / 32) / 8;
  gemm_nt<<<gblocks, 256, 0, stream>>>(qx, wqb, bq, Qb, nullptr,
                                       S_LEN, E_DIM, E_DIM);
  gemm_nt<<<gblocks, 256, 0, stream>>>(kx, wkb, bk, Kb, nullptr,
                                       S_LEN, E_DIM, E_DIM);
  gemm_nt<<<gblocks, 256, 0, stream>>>(vx, wvb, bv, Vb, nullptr,
                                       S_LEN, E_DIM, E_DIM);

  // fused top-k attention: 16 heads * 128 q-blocks
  const size_t smem = (size_t)KSTAGE_BYTES + (size_t)16 * S_LEN * 4 +
                      (size_t)16 * TOPK * 4 * 3;
  attn_topk_kernel<<<H_NUM * (S_LEN / 16), 256, smem, stream>>>(
      Qb, Kb, Vb, ctx, amean);

  // output projection into d_out (f32)
  gemm_nt<<<gblocks, 256, 0, stream>>>(ctx, wob, bo, nullptr, out,
                                       S_LEN, E_DIM, E_DIM);
}
